// Raim_22660247454383
// MI455X (gfx1250) — matrix operand order repaired and checked by reading, not compiled, not run on hardware
//
#include <hip/hip_runtime.h>
#include <hip/hip_bf16.h>
#include <math.h>

// ---------------------------------------------------------------------------
// MI455X (gfx1250, wave32) implementation.
// GEMMs use v_wmma_f32_16x16x32_bf16; fragments built with contiguous K per lane half:
//   A (16x32 bf16): lane m = lane&15, K base = (lane>=16 ? 16 : 0),
//                   element h -> K = base + h
//   B (32x16 bf16): lane n = lane&15, K = (lane>=16 ? 16 : 0) + h
//   C/D (16x16 f32): element r -> (M = r + (lane>=16 ? 8 : 0), N = lane&15)
// Embed kernel stages A-tiles with the Tensor Data Mover (double-buffered,
// s_wait_tensorcnt handoff) when the builtin is available.
// ---------------------------------------------------------------------------

typedef __bf16 bf16;
typedef __attribute__((ext_vector_type(16))) __bf16 v16bf;
typedef __attribute__((ext_vector_type(8)))  float  v8f;
typedef __attribute__((ext_vector_type(4)))  unsigned int v4u;
typedef __attribute__((ext_vector_type(8)))  int v8i;
typedef __attribute__((ext_vector_type(4)))  int v4i;

#define Bsz 512
#define Lsz 48
#define Dsz 256
#define Vsz 512
#define V3sz 64

#if defined(__AMDGCN__) && defined(__has_builtin)
#if __has_builtin(__builtin_amdgcn_tensor_load_to_lds) && \
    __has_builtin(__builtin_amdgcn_s_wait_tensorcnt)
#define USE_TDM 1
#endif
#endif

__device__ __forceinline__ v8f v8zero() {
  v8f z = {0.f, 0.f, 0.f, 0.f, 0.f, 0.f, 0.f, 0.f};
  return z;
}

__device__ __forceinline__ v8f wmma_bf16(v16bf a, v16bf b, v8f c) {
  return __builtin_amdgcn_wmma_f32_16x16x32_bf16(false, a, false, b,
                                                 (short)0, c, false, false);
}

// A fragment (16x32) from LDS bf16 tile, row-major [16][stride]
__device__ __forceinline__ v16bf ldsA_frag(const bf16* tile, int stride,
                                           int lane, int k0) {
  const int m  = lane & 15;
  const int kb = (lane & 16) ? 16 : 0;
  v16bf f;
#pragma unroll
  for (int h = 0; h < 16; ++h) {
    int k = k0 + kb + h;
    f[h] = tile[m * stride + k];
  }
  return f;
}

// A fragment (16x32) from LDS f32 tile (TDM-staged), convert while gathering
__device__ __forceinline__ v16bf ldsA_frag_f32(const float* tile, int stride,
                                               int lane, int k0) {
  const int m  = lane & 15;
  const int kb = (lane & 16) ? 16 : 0;
  v16bf f;
#pragma unroll
  for (int h = 0; h < 16; ++h) {
    int k = k0 + kb + h;
    f[h] = (bf16)tile[m * stride + k];
  }
  return f;
}

// B fragment (32x16) from LDS bf16 stored [n][stride]
__device__ __forceinline__ v16bf ldsB_frag(const bf16* tile, int stride,
                                           int lane, int k0) {
  const int n  = lane & 15;
  const int kb = (lane & 16) ? 16 : 0;
  v16bf f;
#pragma unroll
  for (int h = 0; h < 16; ++h) f[h] = tile[n * stride + k0 + kb + h];
  return f;
}

// B fragment (32x16) from global bf16 weight [rows][ld] — one aligned 32-byte
// vector load per lane (16 contiguous bf16 along K).
__device__ __forceinline__ v16bf gmemB_frag_bf16(const bf16* __restrict__ w,
                                                 int ld, int row0, int lane,
                                                 int k0) {
  const int n  = lane & 15;
  const int kb = (lane & 16) ? 16 : 0;
  return *(const v16bf*)(w + (size_t)(row0 + n) * ld + k0 + kb);
}

#if USE_TDM
// Issue a TDM 2D tile load: tile_h rows x tile_w f32 elements, row stride
// row_stride elements, global->LDS (contiguous row-major in LDS).
__device__ __forceinline__ void tdm_load_tile_f32(const float* gsrc, void* ldst,
                                                  unsigned tile_w,
                                                  unsigned tile_h,
                                                  unsigned row_stride) {
  unsigned lds_off = (unsigned)(size_t)ldst;
  unsigned long long ga = (unsigned long long)(size_t)gsrc;
  // D# group0: count=1 | lds_addr | global_addr[56:0] | type=2 ("image")
  v4u g0 = {1u, lds_off, (unsigned)ga,
            ((unsigned)(ga >> 32) & 0x1FFFFFFu) | (2u << 30)};
  // D# group1: data_size=4B; tensor_dim0=row_stride; tensor_dim1=1<<20;
  //            tile_dim0=tile_w; tile_dim1=tile_h; dim0_stride=row_stride
  int w1 = (int)((row_stride & 0xFFFFu) << 16);
  int w2 = (int)(row_stride >> 16);
  int w3 = (int)(0x10u | (tile_w << 16));
  int w4 = (int)(tile_h & 0xFFFFu);
  v8i g1 = {(int)(2u << 16), w1, w2, w3, w4, (int)row_stride, 0, 0};
  v4i gz = {0, 0, 0, 0};
#if __clang_major__ >= 23
  v8i gz8 = {0, 0, 0, 0, 0, 0, 0, 0};
  __builtin_amdgcn_tensor_load_to_lds(g0, g1, gz, gz, gz8, 0);
#else
  __builtin_amdgcn_tensor_load_to_lds(g0, g1, gz, gz, 0);
#endif
}
#endif

// ---------------------------------------------------------------------------
// Kernel 0: fp32 -> bf16 weight conversion (one-shot, feeds the recurrence)
// ---------------------------------------------------------------------------
__global__ __launch_bounds__(256) void cvt_bf16_kernel(
    const float* __restrict__ src, bf16* __restrict__ dst, int n) {
  int i = blockIdx.x * 256 + threadIdx.x;
  if (i < n) dst[i] = (bf16)src[i];
}

// ---------------------------------------------------------------------------
// Kernel 1: fused embedding  x = relu(x1@w_e1^T+b1) + relu(x2@w_e2^T+b2)
// also emits x2e channel 0 (post-relu) for the prefix cumsum.
// Block: 64 rows x 64 cols, 8 waves; TDM double-buffers the A tiles.
// ---------------------------------------------------------------------------
__global__ __launch_bounds__(256) void embed_kernel(
    const float* __restrict__ x1, const float* __restrict__ x2,
    const float* __restrict__ w_e1, const float* __restrict__ b_e1,
    const float* __restrict__ w_e2, const float* __restrict__ b_e2,
    float* __restrict__ xo, float* __restrict__ x2e0) {
  __shared__ bf16 lB1[64][32];  // [n][k]
  __shared__ bf16 lB2[64][32];

  const int tid  = threadIdx.x;
  const int wv   = tid >> 5;
  const int lane = tid & 31;
  const int m0 = blockIdx.x * 64;
  const int n0 = blockIdx.y * 64;
  const int mw = wv & 3;   // row tile within block
  const int cw = wv >> 2;  // 0..1 -> col tiles cw*2 + {0,1}

  v8f acc1[2] = {v8zero(), v8zero()};
  v8f acc2[2] = {v8zero(), v8zero()};

#if USE_TDM
  __shared__ float lA1f[2][64][32];
  __shared__ float lA2f[2][64][32];
  if (wv == 0) {  // prologue: kick off k-step 0 DMA
    tdm_load_tile_f32(x1 + (size_t)m0 * Vsz, &lA1f[0][0][0], 32u, 64u, (unsigned)Vsz);
    tdm_load_tile_f32(x2 + (size_t)m0 * Vsz, &lA2f[0][0][0], 32u, 64u, (unsigned)Vsz);
  }
  for (int ks = 0; ks < 16; ++ks) {
    const int k0 = ks * 32;
    const int cur = ks & 1, nxt = cur ^ 1;
    __syncthreads();  // previous iteration's fragment reads complete
    // stage B (weights) cooperatively, converting to bf16
#pragma unroll
    for (int it = 0; it < 8; ++it) {
      int idx = tid + it * 256;
      int r = idx >> 5, c = idx & 31;
      lB1[r][c] = (bf16)w_e1[(size_t)(n0 + r) * Vsz + k0 + c];
      lB2[r][c] = (bf16)w_e2[(size_t)(n0 + r) * Vsz + k0 + c];
    }
    if ((k0 + 32 < Vsz) && tid < 64) {  // prefetch next weight rows into L2
      __builtin_prefetch(&w_e1[(size_t)(n0 + tid) * Vsz + k0 + 32], 0, 3);
      __builtin_prefetch(&w_e2[(size_t)(n0 + tid) * Vsz + k0 + 32], 0, 3);
    }
    if (wv == 0) {
      __builtin_amdgcn_s_wait_tensorcnt(0);  // A[cur] landed in LDS
      if (ks + 1 < 16) {                     // issue A[next] DMA
        tdm_load_tile_f32(x1 + (size_t)m0 * Vsz + (k0 + 32), &lA1f[nxt][0][0],
                          32u, 64u, (unsigned)Vsz);
        tdm_load_tile_f32(x2 + (size_t)m0 * Vsz + (k0 + 32), &lA2f[nxt][0][0],
                          32u, 64u, (unsigned)Vsz);
      }
    }
    __syncthreads();
    v16bf a1 = ldsA_frag_f32(&lA1f[cur][mw * 16][0], 32, lane, 0);
    v16bf a2 = ldsA_frag_f32(&lA2f[cur][mw * 16][0], 32, lane, 0);
#pragma unroll
    for (int ct = 0; ct < 2; ++ct) {
      int nb = (cw * 2 + ct) * 16;
      v16bf bf1 = ldsB_frag(&lB1[nb][0], 32, lane, 0);
      v16bf bf2 = ldsB_frag(&lB2[nb][0], 32, lane, 0);
      acc1[ct] = wmma_bf16(a1, bf1, acc1[ct]);
      acc2[ct] = wmma_bf16(a2, bf2, acc2[ct]);
    }
  }
#else
  __shared__ bf16 lA1[64][32];
  __shared__ bf16 lA2[64][32];
  for (int k0 = 0; k0 < Vsz; k0 += 32) {
    __syncthreads();
#pragma unroll
    for (int it = 0; it < 8; ++it) {
      int idx = tid + it * 256;
      int r = idx >> 5, c = idx & 31;
      lA1[r][c] = (bf16)x1[(size_t)(m0 + r) * Vsz + k0 + c];
      lA2[r][c] = (bf16)x2[(size_t)(m0 + r) * Vsz + k0 + c];
      lB1[r][c] = (bf16)w_e1[(size_t)(n0 + r) * Vsz + k0 + c];
      lB2[r][c] = (bf16)w_e2[(size_t)(n0 + r) * Vsz + k0 + c];
    }
    __syncthreads();
    v16bf a1 = ldsA_frag(&lA1[mw * 16][0], 32, lane, 0);
    v16bf a2 = ldsA_frag(&lA2[mw * 16][0], 32, lane, 0);
#pragma unroll
    for (int ct = 0; ct < 2; ++ct) {
      int nb = (cw * 2 + ct) * 16;
      v16bf bf1 = ldsB_frag(&lB1[nb][0], 32, lane, 0);
      v16bf bf2 = ldsB_frag(&lB2[nb][0], 32, lane, 0);
      acc1[ct] = wmma_bf16(a1, bf1, acc1[ct]);
      acc2[ct] = wmma_bf16(a2, bf2, acc2[ct]);
    }
  }
#endif

  const int n  = lane & 15;
  const int mh = (lane & 16) ? 8 : 0;
#pragma unroll
  for (int ct = 0; ct < 2; ++ct) {
    int col = n0 + (cw * 2 + ct) * 16 + n;
    float bb1 = b_e1[col], bb2 = b_e2[col];
#pragma unroll
    for (int r = 0; r < 8; ++r) {
      int row = m0 + mw * 16 + mh + r;
      float v1 = acc1[ct][r] + bb1; v1 = v1 > 0.f ? v1 : 0.f;
      float v2 = acc2[ct][r] + bb2; v2 = v2 > 0.f ? v2 : 0.f;
      xo[(size_t)row * Dsz + col] = v1 + v2;
      if (col == 0) x2e0[row] = v2;
    }
  }
}

// ---------------------------------------------------------------------------
// Kernel 2a: xw[b,l] = dot(x[b,l,:], w_x)  (step-invariant part of t1)
// ---------------------------------------------------------------------------
__global__ __launch_bounds__(256) void xw_kernel(const float* __restrict__ x,
                                                 const float* __restrict__ w_x,
                                                 float* __restrict__ xw) {
  const int wv = threadIdx.x >> 5, lane = threadIdx.x & 31;
  const int row = blockIdx.x * 8 + wv;
  float acc = 0.f;
#pragma unroll
  for (int j = 0; j < 8; ++j) {
    int d = lane + 32 * j;
    acc += x[(size_t)row * Dsz + d] * w_x[d];
  }
#pragma unroll
  for (int off = 16; off > 0; off >>= 1) acc += __shfl_down(acc, off, 32);
  if (lane == 0) xw[row] = acc;
}

// ---------------------------------------------------------------------------
// Kernel 2b: prefix[b,l] = sum_{j<l} x2e0[b,j]
// ---------------------------------------------------------------------------
__global__ __launch_bounds__(256) void prefix_kernel(
    const float* __restrict__ x2e0, float* __restrict__ prefix) {
  int b = blockIdx.x * blockDim.x + threadIdx.x;
  if (b >= Bsz) return;
  float run = 0.f;
  for (int l = 0; l < Lsz; ++l) {
    prefix[(size_t)b * Lsz + l] = run;
    run += x2e0[(size_t)b * Lsz + l];
  }
}

// ---------------------------------------------------------------------------
// Kernel 3: full 48-step recurrence. Block owns 16 batches; h ping-pongs in
// LDS; GEMMs via WMMA bf16 with pre-converted bf16 weights.
// ---------------------------------------------------------------------------
__global__ __launch_bounds__(256) void recur_kernel(
    const float* __restrict__ x, const float* __restrict__ xw,
    const float* __restrict__ prefix,
    const bf16* __restrict__ whb, const float* __restrict__ b_h,
    const bf16* __restrict__ wihb, const bf16* __restrict__ whhb,
    const float* __restrict__ b_ih, const float* __restrict__ b_hh,
    float* __restrict__ hF) {
  __shared__ float h_buf[2][16][Dsz];   // ping-pong hidden state
  __shared__ bf16  hbf[16][Dsz];        // bf16 copy of h (GEMM A)
  __shared__ bf16  ctxbf[16][Dsz];      // bf16 context (GEMM A)
  __shared__ float t1s[16][Lsz];
  __shared__ float a_s[16][Lsz];
  __shared__ float xw_s[16][Lsz];
  __shared__ int   conds[16];

  const int tid  = threadIdx.x;
  const int wv   = tid >> 5;
  const int lane = tid & 31;
  const int batch0 = blockIdx.x * 16;

  for (int idx = tid; idx < 16 * Dsz; idx += 256) (&h_buf[0][0][0])[idx] = 0.f;
  for (int idx = tid; idx < 16 * Lsz; idx += 256) {
    int b = idx / Lsz, l = idx % Lsz;
    xw_s[b][l] = xw[(size_t)(batch0 + b) * Lsz + l];
  }

  for (int i = 0; i < Lsz; ++i) {
    const int src = i & 1, dst = src ^ 1;
    __syncthreads();
    for (int idx = tid; idx < 16 * Dsz; idx += 256)
      (&hbf[0][0])[idx] = (bf16)(&h_buf[src][0][0])[idx];
    if (tid < 16)
      conds[tid] = (prefix[(size_t)(batch0 + tid) * Lsz + i] == 0.f) ? 1 : 0;
    __syncthreads();

    // ---- t1 = h @ w_h^T  (M=16, N=48 -> 3 tiles over waves 0..2) ----
    if (wv < 3) {
      v8f acc = v8zero();
      for (int ks = 0; ks < 8; ++ks) {
        v16bf a  = ldsA_frag(&hbf[0][0], Dsz, lane, ks * 32);
        v16bf bf = gmemB_frag_bf16(whb, Dsz, wv * 16, lane, ks * 32);
        acc = wmma_bf16(a, bf, acc);
      }
      const int n = lane & 15, mh = (lane & 16) ? 8 : 0;
#pragma unroll
      for (int r = 0; r < 8; ++r) t1s[mh + r][wv * 16 + n] = acc[r];
    }
    __syncthreads();

    // ---- finalize t1, softmax over L, apply l<=i mask -> a_s ----
    if (tid < 16) {
      const int b = tid;
      float mx = -1e30f;
      for (int l = 0; l < Lsz; ++l) {
        int g = (l == i) || (conds[b] && (l < i));
        float t = t1s[b][l] + b_h[l] + (g ? xw_s[b][l] : 0.f);
        t1s[b][l] = t;
        mx = t > mx ? t : mx;
      }
      float sum = 0.f;
      for (int l = 0; l < Lsz; ++l) {
        float e = __expf(t1s[b][l] - mx);
        t1s[b][l] = e;
        sum += e;
      }
      float inv = 1.f / sum;
      for (int l = 0; l < Lsz; ++l)
        a_s[b][l] = (l <= i) ? t1s[b][l] * inv : 0.f;
    }
    __syncthreads();

    // ---- ctx[b,d] = sum_{l<=i} a[b,l]*x[b,l,d] (per-batch, VALU) ----
    {
      const int b = tid >> 4, dl = tid & 15;
      for (int j = 0; j < 16; ++j) {
        int d = dl + 16 * j;
        float acc = 0.f;
        for (int l = 0; l <= i; ++l)
          acc += a_s[b][l] * x[((size_t)(batch0 + b) * Lsz + l) * Dsz + d];
        ctxbf[b][d] = (bf16)acc;
      }
    }
    __syncthreads();

    // ---- GRU gates: gi = ctx@w_ih^T, gh = h@w_hh^T ; wave owns 2 d-chunks --
    for (int cc = 0; cc < 2; ++cc) {
      const int dbase = (wv + cc * 8) * 16;
      v8f gi_r = v8zero(), gi_z = v8zero(), gi_n = v8zero();
      v8f gh_r = v8zero(), gh_z = v8zero(), gh_n = v8zero();
      for (int ks = 0; ks < 8; ++ks) {
        int k0 = ks * 32;
        v16bf ac = ldsA_frag(&ctxbf[0][0], Dsz, lane, k0);
        v16bf ah = ldsA_frag(&hbf[0][0], Dsz, lane, k0);
        v16bf wi0 = gmemB_frag_bf16(wihb, Dsz, 0 * Dsz + dbase, lane, k0);
        v16bf wi1 = gmemB_frag_bf16(wihb, Dsz, 1 * Dsz + dbase, lane, k0);
        v16bf wi2 = gmemB_frag_bf16(wihb, Dsz, 2 * Dsz + dbase, lane, k0);
        v16bf wh0 = gmemB_frag_bf16(whhb, Dsz, 0 * Dsz + dbase, lane, k0);
        v16bf wh1 = gmemB_frag_bf16(whhb, Dsz, 1 * Dsz + dbase, lane, k0);
        v16bf wh2 = gmemB_frag_bf16(whhb, Dsz, 2 * Dsz + dbase, lane, k0);
        gi_r = wmma_bf16(ac, wi0, gi_r);
        gi_z = wmma_bf16(ac, wi1, gi_z);
        gi_n = wmma_bf16(ac, wi2, gi_n);
        gh_r = wmma_bf16(ah, wh0, gh_r);
        gh_z = wmma_bf16(ah, wh1, gh_z);
        gh_n = wmma_bf16(ah, wh2, gh_n);
      }
      const int n = lane & 15, mh = (lane & 16) ? 8 : 0;
      const int d = dbase + n;
#pragma unroll
      for (int r = 0; r < 8; ++r) {
        int m = mh + r;
        float rr = 1.f / (1.f + __expf(-(gi_r[r] + b_ih[d] + gh_r[r] + b_hh[d])));
        float zz = 1.f / (1.f + __expf(-(gi_z[r] + b_ih[Dsz + d] + gh_z[r] + b_hh[Dsz + d])));
        float nn = tanhf(gi_n[r] + b_ih[2 * Dsz + d] + rr * (gh_n[r] + b_hh[2 * Dsz + d]));
        h_buf[dst][m][d] = (1.f - zz) * nn + zz * h_buf[src][m][d];
      }
    }
  }
  __syncthreads();
  for (int idx = tid; idx < 16 * Dsz; idx += 256) {
    int b = idx >> 8, d = idx & 255;
    hF[(size_t)(batch0 + b) * Dsz + d] = h_buf[0][b][d];  // 48 steps end in buf 0
  }
}

// ---------------------------------------------------------------------------
// Kernel 4: y[b] = sigmoid(sum_d (hF+se)[b,d]*w_out[d] + b_out)
// ---------------------------------------------------------------------------
__global__ __launch_bounds__(256) void out_kernel(
    const float* __restrict__ hF, const float* __restrict__ s,
    const float* __restrict__ w_s, const float* __restrict__ b_s,
    const float* __restrict__ w_out, const float* __restrict__ b_out,
    float* __restrict__ y) {
  const int wv = threadIdx.x >> 5, lane = threadIdx.x & 31;
  const int b = blockIdx.x * 8 + wv;
  if (b >= Bsz) return;
  float acc = 0.f;
#pragma unroll
  for (int j = 0; j < 8; ++j) {
    int d = lane + 32 * j;
    float se = b_s[d];
    for (int v = 0; v < V3sz; ++v)
      se += s[(size_t)b * V3sz + v] * w_s[(size_t)d * V3sz + v];
    acc += (hF[(size_t)b * Dsz + d] + se) * w_out[d];
  }
#pragma unroll
  for (int off = 16; off > 0; off >>= 1) acc += __shfl_down(acc, off, 32);
  if (lane == 0) y[b] = 1.f / (1.f + __expf(-(acc + b_out[0])));
}

// ---------------------------------------------------------------------------
extern "C" void kernel_launch(void* const* d_in, const int* in_sizes, int n_in,
                              void* d_out, int out_size, void* d_ws,
                              size_t ws_size, hipStream_t stream) {
  const float* x1   = (const float*)d_in[0];
  const float* x2   = (const float*)d_in[1];
  const float* s    = (const float*)d_in[2];
  const float* w_e1 = (const float*)d_in[3];
  const float* b_e1 = (const float*)d_in[4];
  const float* w_e2 = (const float*)d_in[5];
  const float* b_e2 = (const float*)d_in[6];
  const float* w_s  = (const float*)d_in[7];
  const float* b_s  = (const float*)d_in[8];
  const float* w_x  = (const float*)d_in[9];
  const float* w_h  = (const float*)d_in[10];
  const float* b_h  = (const float*)d_in[11];
  const float* w_ih = (const float*)d_in[12];
  const float* w_hh = (const float*)d_in[13];
  const float* b_ih = (const float*)d_in[14];
  const float* b_hh = (const float*)d_in[15];
  const float* w_out = (const float*)d_in[16];
  const float* b_out = (const float*)d_in[17];

  float* ws       = (float*)d_ws;
  float* ws_x     = ws;                                  // B*L*D f32
  float* ws_xw    = ws_x + (size_t)Bsz * Lsz * Dsz;      // B*L
  float* ws_x2e0  = ws_xw + (size_t)Bsz * Lsz;           // B*L
  float* ws_pref  = ws_x2e0 + (size_t)Bsz * Lsz;         // B*L
  float* ws_hF    = ws_pref + (size_t)Bsz * Lsz;         // B*D
  bf16*  ws_whb   = (bf16*)(ws_hF + (size_t)Bsz * Dsz);  // L*D bf16
  bf16*  ws_wihb  = ws_whb + (size_t)Lsz * Dsz;          // 3D*D bf16
  bf16*  ws_whhb  = ws_wihb + (size_t)3 * Dsz * Dsz;     // 3D*D bf16

  cvt_bf16_kernel<<<(Lsz * Dsz + 255) / 256, 256, 0, stream>>>(w_h, ws_whb,
                                                               Lsz * Dsz);
  cvt_bf16_kernel<<<(3 * Dsz * Dsz + 255) / 256, 256, 0, stream>>>(
      w_ih, ws_wihb, 3 * Dsz * Dsz);
  cvt_bf16_kernel<<<(3 * Dsz * Dsz + 255) / 256, 256, 0, stream>>>(
      w_hh, ws_whhb, 3 * Dsz * Dsz);

  embed_kernel<<<dim3((Bsz * Lsz) / 64, Dsz / 64), 256, 0, stream>>>(
      x1, x2, w_e1, b_e1, w_e2, b_e2, ws_x, ws_x2e0);
  xw_kernel<<<(Bsz * Lsz) / 8, 256, 0, stream>>>(ws_x, w_x, ws_xw);
  prefix_kernel<<<(Bsz + 255) / 256, 256, 0, stream>>>(ws_x2e0, ws_pref);
  recur_kernel<<<Bsz / 16, 256, 0, stream>>>(ws_x, ws_xw, ws_pref, ws_whb, b_h,
                                             ws_wihb, ws_whhb, b_ih, b_hh,
                                             ws_hF);
  out_kernel<<<Bsz / 8, 256, 0, stream>>>(ws_hF, s, w_s, b_s, w_out, b_out,
                                          (float*)d_out);
}
